// RegressorBranchless_47055661695451
// MI455X (gfx1250) — matrix operand order repaired and checked by reading, not compiled, not run on hardware
//
#include <hip/hip_runtime.h>
#include <hip/hip_bf16.h>
#include <stdint.h>

// Problem constants (match reference)
#define H_DIM 448
#define W_DIM 608
#define CIN   64
#define D0    128
#define D1    128
#define D2    1024
#define NT    32          // columns (w) per block tile
#define NTILES (W_DIM/16) // 38 column tiles of 16
#define BLOCK 256         // 8 wave32 waves

typedef __attribute__((ext_vector_type(16))) _Float16 v16h;
typedef __attribute__((ext_vector_type(8)))  _Float16 v8h;
typedef __attribute__((ext_vector_type(4)))  _Float16 v4h;
typedef __attribute__((ext_vector_type(8)))  float    v8f;

__device__ __forceinline__ float lrelu(float v) { return v > 0.f ? v : 0.01f * v; }

// ---------------- CDNA5 async global->LDS copy (ASYNCcnt path) ----------------
__device__ __forceinline__ void async_ld_b32(void* lds_dst, const void* gsrc) {
  asm volatile("global_load_async_to_lds_b32 %0, %1, off"
               :
               : "v"((unsigned)(unsigned long long)(uintptr_t)lds_dst),
                 "v"((unsigned long long)(uintptr_t)gsrc)
               : "memory");
}
__device__ __forceinline__ void async_wait0() {
  asm volatile("s_wait_asynccnt 0" ::: "memory");
}

// ---------------- WMMA fragment loaders ---------------------------------------
// A matrix 16x32 f16 (MxK) from row-major LDS: two contiguous 16B runs per lane.
__device__ __forceinline__ v16h load_frag_a(const _Float16* s, int ld) {
  const int lane = threadIdx.x & 31;
  const int m = lane & 15, hi = lane >> 4;
  const _Float16* row = s + m * ld + 8 * hi;
  v8h lo = *(const v8h*)(row);
  v8h up = *(const v8h*)(row + 16);
  return __builtin_shufflevector(lo, up, 0, 1, 2, 3, 4, 5, 6, 7,
                                 8, 9, 10, 11, 12, 13, 14, 15);
}
// B matrix 32x16 f16 (KxN) from COLUMN-MAJOR LDS: two contiguous 16B runs per lane.
__device__ __forceinline__ v16h load_frag_b_cm(const _Float16* s, int ldk) {
  const int lane = threadIdx.x & 31;
  const int n = lane & 15, hi = lane >> 4;
  const _Float16* col = s + n * ldk + 8 * hi;
  v8h lo = *(const v8h*)(col);
  v8h up = *(const v8h*)(col + 16);
  return __builtin_shufflevector(lo, up, 0, 1, 2, 3, 4, 5, 6, 7,
                                 8, 9, 10, 11, 12, 13, 14, 15);
}

// =============================================================================
// Kernel 0: one-shot weight packing to f16.
//  w1p[tap][co][ci]  (tap-major so each tap slice is a contiguous 16 KB blob)
//  w2p[h][d][c], w3p[h][d][c] : straight f32->f16 (w3p = 117 MB, L2-resident)
// =============================================================================
__global__ __launch_bounds__(BLOCK)
void pack_weights(const float* __restrict__ w1, const float* __restrict__ w2,
                  const float* __restrict__ w3, _Float16* __restrict__ w1p,
                  _Float16* __restrict__ w2p, _Float16* __restrict__ w3p) {
  const size_t gt  = (size_t)blockIdx.x * BLOCK + threadIdx.x;
  const size_t gsz = (size_t)gridDim.x * BLOCK;
  for (size_t i = gt; i < 9 * D0 * CIN; i += gsz) {
    const size_t tap = i / (D0 * CIN);
    const size_t rem = i - tap * (D0 * CIN);     // co*64+ci
    w1p[i] = (_Float16)w1[rem * 9 + tap];
  }
  const float4* s2 = (const float4*)w2;
  v4h* d2 = (v4h*)w2p;
  for (size_t i = gt; i < (size_t)H_DIM * D1 * D0 / 4; i += gsz) {
    const float4 f = s2[i];
    v4h p; p[0] = (_Float16)f.x; p[1] = (_Float16)f.y;
    p[2] = (_Float16)f.z; p[3] = (_Float16)f.w;
    d2[i] = p;
  }
  const float4* s3 = (const float4*)w3;
  v4h* d3 = (v4h*)w3p;
  for (size_t i = gt; i < (size_t)H_DIM * D2 * D1 / 4; i += gsz) {
    const float4 f = s3[i];
    v4h p; p[0] = (_Float16)f.x; p[1] = (_Float16)f.y;
    p[2] = (_Float16)f.z; p[3] = (_Float16)f.w;
    d3[i] = p;
  }
}

// =============================================================================
// Kernel 1: 3x3 SAME conv (64->128) as 9-tap implicit GEMM + bias + lrelu.
// Double-buffered pipeline: async-DMA tap t+1 weights (contiguous 16 KB) and
// stage padded x tile for t+1 while WMMAs run on tap t.
// Output y1 packed column-major per 16-wide tile: y1[h][t][n][c], c contiguous.
// =============================================================================
__global__ __launch_bounds__(BLOCK)
void conv3x3_wmma(const float* __restrict__ x, const _Float16* __restrict__ w1p,
                  const float* __restrict__ b1, _Float16* __restrict__ y1) {
  __shared__ alignas(64) _Float16 smW[2][D0 * CIN];  // 2 x 16 KB
  __shared__ alignas(64) _Float16 smX[2][NT * CIN];  // 2 x 4 KB, column-major

  const int w0  = blockIdx.x * NT;
  const int h   = blockIdx.y;
  const int tid = threadIdx.x;
  const int wave = tid >> 5;

  auto stage_w_async = [&](int tap, int buf) {
    const char* src = (const char*)&w1p[(size_t)tap * D0 * CIN];
    for (int d = tid; d < (D0 * CIN) / 2; d += BLOCK)
      async_ld_b32((char*)smW[buf] + (size_t)d * 4, src + (size_t)d * 4);
  };
  // Zero padding via clamped (always in-bounds) loads + select: keeps the
  // global loads unconditional so the compiler can clause them (no per-element
  // exec-divergent load + s_wait serialization).
  auto stage_x = [&](int tap, int buf) {
    const int kh = tap / 3, kw = tap % 3;
    const int hh = h + kh - 1;
    const int hc = min(max(hh, 0), H_DIM - 1);
    const bool hok = (hh >= 0) & (hh < H_DIM);
    for (int i = tid; i < CIN * NT; i += BLOCK) {
      const int ci = i >> 5;    // coalesced along w in global
      const int j  = i & 31;
      const int ww = w0 + kw - 1 + j;
      const int wc = min(max(ww, 0), W_DIM - 1);
      const float raw = x[((size_t)ci * H_DIM + hc) * W_DIM + wc];
      const bool ok = hok & (ww >= 0) & (ww < W_DIM);
      smX[buf][j * CIN + ci] = (_Float16)(ok ? raw : 0.f);
    }
  };

  v8f acc0 = {}; v8f acc1 = {};
  stage_w_async(0, 0);
  stage_x(0, 0);
  async_wait0();
  __syncthreads();

#pragma unroll 1
  for (int tap = 0; tap < 9; ++tap) {
    const int cur = tap & 1;
    if (tap < 8) {                 // overlap DMA + staging with compute
      stage_w_async(tap + 1, cur ^ 1);
      stage_x(tap + 1, cur ^ 1);
    }
    v16h af[2], bl[2], bh[2];
#pragma unroll
    for (int kc = 0; kc < 2; ++kc) {
      af[kc] = load_frag_a(&smW[cur][(wave * 16) * CIN + kc * 32], CIN);
      bl[kc] = load_frag_b_cm(&smX[cur][0 * CIN + kc * 32],  CIN);
      bh[kc] = load_frag_b_cm(&smX[cur][16 * CIN + kc * 32], CIN);
    }
#pragma unroll
    for (int kc = 0; kc < 2; ++kc) {
      acc0 = __builtin_amdgcn_wmma_f32_16x16x32_f16(false, af[kc], false, bl[kc], (short)0, acc0, false, false);
      acc1 = __builtin_amdgcn_wmma_f32_16x16x32_f16(false, af[kc], false, bh[kc], (short)0, acc1, false, false);
    }
    if (tap < 8) async_wait0();
    __syncthreads();
  }

  // Epilogue: bias + lrelu, packed 16B stores (8 consecutive m per half-wave).
  const int lane = tid & 31, n = lane & 15, hi = lane >> 4;
  const size_t tile = (size_t)h * NTILES + blockIdx.x * 2;
  v8h q0, q1;
#pragma unroll
  for (int r = 0; r < 8; ++r) {
    const int m = wave * 16 + r + 8 * hi;
    const float bias = b1[m];
    q0[r] = (_Float16)lrelu(acc0[r] + bias);
    q1[r] = (_Float16)lrelu(acc1[r] + bias);
  }
  *(v8h*)&y1[((tile + 0) * 16 + n) * D0 + wave * 16 + 8 * hi] = q0;
  *(v8h*)&y1[((tile + 1) * 16 + n) * D0 + wave * 16 + 8 * hi] = q1;
}

// =============================================================================
// Kernel 2: per-h GEMM  y2[h] = lrelu(w2p[h] (128x128) @ y1[h] (128x608) + b2)
// Both operands staged with contiguous async DMA (32 KB + 8 KB), zero cvt.
// =============================================================================
__global__ __launch_bounds__(BLOCK)
void gemm_l2(const _Float16* __restrict__ y1, const _Float16* __restrict__ w2p,
             const float* __restrict__ b2, _Float16* __restrict__ y2) {
  __shared__ alignas(64) _Float16 smA[D1 * D0];  // 32 KB, row-major [d][c]
  __shared__ alignas(64) _Float16 smB[NT * D0];  // 8 KB, column-major [col][c]
  const int h   = blockIdx.y;
  const int tid = threadIdx.x;
  const int wave = tid >> 5;
  const size_t tile = (size_t)h * NTILES + blockIdx.x * 2;

  const char* As = (const char*)&w2p[(size_t)h * D1 * D0];
  for (int d = tid; d < (D1 * D0) / 2; d += BLOCK)
    async_ld_b32((char*)smA + (size_t)d * 4, As + (size_t)d * 4);
  const char* Bs = (const char*)&y1[tile * 16 * D0];
  for (int d = tid; d < (NT * D0) / 2; d += BLOCK)
    async_ld_b32((char*)smB + (size_t)d * 4, Bs + (size_t)d * 4);
  async_wait0();
  __syncthreads();

  v16h af[4], bl[4], bh[4];
#pragma unroll
  for (int kc = 0; kc < 4; ++kc) {
    af[kc] = load_frag_a(&smA[(wave * 16) * D0 + kc * 32], D0);
    bl[kc] = load_frag_b_cm(&smB[0 * D0 + kc * 32],  D0);
    bh[kc] = load_frag_b_cm(&smB[16 * D0 + kc * 32], D0);
  }
  v8f acc0 = {}, acc1 = {};
#pragma unroll
  for (int kc = 0; kc < 4; ++kc) {
    acc0 = __builtin_amdgcn_wmma_f32_16x16x32_f16(false, af[kc], false, bl[kc], (short)0, acc0, false, false);
    acc1 = __builtin_amdgcn_wmma_f32_16x16x32_f16(false, af[kc], false, bh[kc], (short)0, acc1, false, false);
  }
  const int lane = tid & 31, n = lane & 15, hi = lane >> 4;
  v8h q0, q1;
#pragma unroll
  for (int r = 0; r < 8; ++r) {
    const int m = wave * 16 + r + 8 * hi;
    const float bias = b2[(size_t)h * D1 + m];
    q0[r] = (_Float16)lrelu(acc0[r] + bias);
    q1[r] = (_Float16)lrelu(acc1[r] + bias);
  }
  *(v8h*)&y2[((tile + 0) * 16 + n) * D1 + wave * 16 + 8 * hi] = q0;
  *(v8h*)&y2[((tile + 1) * 16 + n) * D1 + wave * 16 + 8 * hi] = q1;
}

// =============================================================================
// Kernel 3: fused L3+L4 with double-buffered async slice pipeline.
//   out[h,j,w] = sum_m w4[h,j,m] * lrelu( (w3[h] @ y2[h])[m,w] + b3[h,m] ) + b4
// Never materializes the 1.1 GB y3 tensor; w3p slices (32 KB) DMA'd while
// the previous slice's WMMAs execute.
// =============================================================================
__global__ __launch_bounds__(BLOCK)
void gemm_l3l4(const _Float16* __restrict__ y2, const _Float16* __restrict__ w3p,
               const float* __restrict__ b3, const float* __restrict__ w4,
               const float* __restrict__ b4, float* __restrict__ out) {
  __shared__ alignas(64) _Float16 smA[2][128 * D1];  // 2 x 32 KB: w3 slices
  __shared__ alignas(64) _Float16 smB[NT * D1];      // 8 KB: y2 tile, col-major
  __shared__ float smW4[2 * D2];                     // 8 KB
  __shared__ float smB3[D2];                         // 4 KB
  __shared__ float smOut[2 * NT];

  const int w0  = blockIdx.x * NT;
  const int h   = blockIdx.y;
  const int tid = threadIdx.x;
  const int wave = tid >> 5;
  const int lane = tid & 31, n = lane & 15, hi = lane >> 4;
  const size_t tile = (size_t)h * NTILES + blockIdx.x * 2;
  const _Float16* W3h = &w3p[(size_t)h * D2 * D1];

  auto stage_slice_async = [&](int it, int buf) {
    const char* src = (const char*)&W3h[(size_t)it * 128 * D1];
    for (int d = tid; d < (128 * D1) / 2; d += BLOCK)
      async_ld_b32((char*)smA[buf] + (size_t)d * 4, src + (size_t)d * 4);
  };

  // Stage B (8 KB async), w4 rows, b3 row, zero accumulators, prime slice 0.
  const char* Bs = (const char*)&y2[tile * 16 * D1];
  for (int d = tid; d < (NT * D1) / 2; d += BLOCK)
    async_ld_b32((char*)smB + (size_t)d * 4, Bs + (size_t)d * 4);
  for (int i = tid; i < 2 * D2; i += BLOCK) smW4[i] = w4[(size_t)h * 2 * D2 + i];
  for (int i = tid; i < D2; i += BLOCK)     smB3[i] = b3[(size_t)h * D2 + i];
  if (tid < 2 * NT) smOut[tid] = 0.f;
  stage_slice_async(0, 0);
  async_wait0();
  __syncthreads();

#pragma unroll 1
  for (int it = 0; it < D2 / 128; ++it) {
    const int cur = it & 1;
    if (it + 1 < D2 / 128) stage_slice_async(it + 1, cur ^ 1);  // DMA overlaps WMMA

    v16h af[4], blf[4], bhf[4];
#pragma unroll
    for (int kc = 0; kc < 4; ++kc) {
      af[kc]  = load_frag_a(&smA[cur][(wave * 16) * D1 + kc * 32], D1);
      blf[kc] = load_frag_b_cm(&smB[0 * D1 + kc * 32],  D1);
      bhf[kc] = load_frag_b_cm(&smB[16 * D1 + kc * 32], D1);
    }
    v8f acc0 = {}, acc1 = {};
#pragma unroll
    for (int kc = 0; kc < 4; ++kc) {
      acc0 = __builtin_amdgcn_wmma_f32_16x16x32_f16(false, af[kc], false, blf[kc], (short)0, acc0, false, false);
      acc1 = __builtin_amdgcn_wmma_f32_16x16x32_f16(false, af[kc], false, bhf[kc], (short)0, acc1, false, false);
    }

    // Fused L4 contraction using the C/D layout (M = r + 8*hi per lane).
    const int mbase = it * 128 + wave * 16;
    float p00 = 0.f, p01 = 0.f, p10 = 0.f, p11 = 0.f;
#pragma unroll
    for (int r = 0; r < 8; ++r) {
      const int m = mbase + r + 8 * hi;
      const float a0 = lrelu(acc0[r] + smB3[m]);
      const float a1 = lrelu(acc1[r] + smB3[m]);
      const float wj0 = smW4[m], wj1 = smW4[D2 + m];
      p00 += wj0 * a0; p10 += wj1 * a0;
      p01 += wj0 * a1; p11 += wj1 * a1;
    }
    p00 += __shfl_xor(p00, 16, 32);
    p01 += __shfl_xor(p01, 16, 32);
    p10 += __shfl_xor(p10, 16, 32);
    p11 += __shfl_xor(p11, 16, 32);
    if (hi == 0) {
      atomicAdd(&smOut[n],           p00);
      atomicAdd(&smOut[16 + n],      p01);
      atomicAdd(&smOut[NT + n],      p10);
      atomicAdd(&smOut[NT + 16 + n], p11);
    }
    if (it + 1 < D2 / 128) async_wait0();
    __syncthreads();
  }

  if (tid < NT) {
    const int w = w0 + tid;
    const float v0 = smOut[tid]      + b4[h * 2 + 0];
    const float v1 = smOut[NT + tid] + b4[h * 2 + 1];
    out[(size_t)h * W_DIM + w] = v0;                                // out
    out[(size_t)H_DIM * W_DIM + (size_t)h * W_DIM + w] = lrelu(v1); // mask
  }
}

extern "C" void kernel_launch(void* const* d_in, const int* in_sizes, int n_in,
                              void* d_out, int out_size, void* d_ws, size_t ws_size,
                              hipStream_t stream) {
  (void)in_sizes; (void)n_in; (void)out_size; (void)ws_size;
  const float* x  = (const float*)d_in[0];
  const float* w1 = (const float*)d_in[1];
  const float* b1 = (const float*)d_in[2];
  const float* w2 = (const float*)d_in[3];
  const float* b2 = (const float*)d_in[4];
  const float* w3 = (const float*)d_in[5];
  const float* b3 = (const float*)d_in[6];
  const float* w4 = (const float*)d_in[7];
  const float* b4 = (const float*)d_in[8];
  float* out = (float*)d_out;

  // Workspace layout (halves): y1, y2 (packed activations), packed weights.
  const size_t y_elems  = (size_t)H_DIM * NTILES * 16 * D0;  // 34,865,152
  const size_t w1_elems = (size_t)9 * D0 * CIN;              // 73,728
  const size_t w2_elems = (size_t)H_DIM * D1 * D0;           // 7,340,032
  _Float16* y1  = (_Float16*)d_ws;
  _Float16* y2  = y1  + y_elems;
  _Float16* w1p = y2  + y_elems;
  _Float16* w2p = w1p + w1_elems;
  _Float16* w3p = w2p + w2_elems;

  dim3 grid(W_DIM / NT, H_DIM, 1);   // 19 x 448 blocks
  dim3 block(BLOCK, 1, 1);           // 8 wave32 waves

  pack_weights<<<dim3(2048, 1, 1), block, 0, stream>>>(w1, w2, w3, w1p, w2p, w3p);
  conv3x3_wmma<<<grid, block, 0, stream>>>(x, w1p, b1, y1);
  gemm_l2<<<grid, block, 0, stream>>>(y1, w2p, b2, y2);
  gemm_l3l4<<<grid, block, 0, stream>>>(y2, w3p, b3, w4, b4, out);
}
